// AttentionBlock_27685359190788
// MI455X (gfx1250) — matrix operand order repaired and checked by reading, not compiled, not run on hardware
//
#include <hip/hip_runtime.h>
#include <hip/hip_bf16.h>
#include <stdint.h>

typedef _Float16 half_t;
typedef __attribute__((ext_vector_type(16))) _Float16 v16h;
typedef __attribute__((ext_vector_type(8)))  float    v8f;

static constexpr int D    = 512;    // d_inner
static constexpr int DH   = 64;
static constexpr int SEQ  = 8192;
static constexpr int MROWS = 4 * SEQ; // 32768

// ---------- small helpers ----------
__device__ __forceinline__ v8f zero8() {
  v8f z;
#pragma unroll
  for (int i = 0; i < 8; ++i) z[i] = 0.0f;
  return z;
}
__device__ __forceinline__ float redmax16(float v) {
#pragma unroll
  for (int m = 1; m < 16; m <<= 1) v = fmaxf(v, __shfl_xor(v, m, 32));
  return v;
}
__device__ __forceinline__ float redsum16(float v) {
#pragma unroll
  for (int m = 1; m < 16; m <<= 1) v += __shfl_xor(v, m, 32);
  return v;
}
__device__ __forceinline__ float redsum32(float v) {
#pragma unroll
  for (int m = 1; m < 32; m <<= 1) v += __shfl_xor(v, m, 32);
  return v;
}

// ---------- CDNA5 async global->LDS staging (ASYNCcnt-tracked DMA) ----------
// lds: byte offset of destination (low 32 bits of flat LDS address)
// gp : per-lane global source address; INST_OFFSET applies to both sides.
__device__ __forceinline__ void async_b128(uint32_t lds, const void* gp) {
  asm volatile("global_load_async_to_lds_b128 %0, %1, off"
               :: "v"(lds), "v"(gp) : "memory");
}
__device__ __forceinline__ void async_b128_off16(uint32_t lds, const void* gp) {
  asm volatile("global_load_async_to_lds_b128 %0, %1, off offset:16"
               :: "v"(lds), "v"(gp) : "memory");
}
__device__ __forceinline__ void async_b128_off32(uint32_t lds, const void* gp) {
  asm volatile("global_load_async_to_lds_b128 %0, %1, off offset:32"
               :: "v"(lds), "v"(gp) : "memory");
}
__device__ __forceinline__ void async_b128_off48(uint32_t lds, const void* gp) {
  asm volatile("global_load_async_to_lds_b128 %0, %1, off offset:48"
               :: "v"(lds), "v"(gp) : "memory");
}
__device__ __forceinline__ void wait_async0() {
  asm volatile("s_wait_asynccnt 0x0" ::: "memory");
}
__device__ __forceinline__ uint32_t lds_addr(const void* p) {
  return (uint32_t)(uintptr_t)p;   // flat->LDS truncates to low 32 bits
}

// ---------- WMMA fragment loaders (wave32, 16x16x32 f16) ----------
// A-matrix 16x32: row M = lane&15; element i -> K = i + 8*(i>=8) + 8*(lane>=16)
__device__ __forceinline__ v16h load_a_frag(const half_t* base, int ld) {
  int lane = threadIdx.x & 31;
  const half_t* rp = base + (size_t)(lane & 15) * ld + ((lane >= 16) ? 8 : 0);
  union { v16h v; uint32_t u[8]; } f;
#pragma unroll
  for (int j = 0; j < 8; ++j) {
    int c = 2 * j + ((2 * j >= 8) ? 8 : 0);
    f.u[j] = *(const uint32_t*)(rp + c);
  }
  return f.v;
}
// B-matrix 32x16 from row-major [N][K]: col N = lane&15; element i -> K = i + 8*(i>=8) + 8*(lane>=16)
__device__ __forceinline__ v16h load_b_frag_nk(const half_t* base, int ld) {
  int lane = threadIdx.x & 31;
  const half_t* rp = base + (size_t)(lane & 15) * ld + ((lane >= 16) ? 8 : 0);
  union { v16h v; uint32_t u[8]; } f;
#pragma unroll
  for (int j = 0; j < 8; ++j) f.u[j] = *(const uint32_t*)(rp + 2 * j + ((2 * j >= 8) ? 8 : 0));
  return f.v;
}
// B-matrix 32x16 from row-major [K][N]
__device__ __forceinline__ v16h load_b_frag_kn(const half_t* base, int ld) {
  int lane = threadIdx.x & 31;
  const half_t* cp = base + (lane & 15) + (size_t)((lane >= 16) ? 8 : 0) * ld;
  union { v16h v; half_t h[16]; } f;
#pragma unroll
  for (int i = 0; i < 16; ++i) f.h[i] = cp[(size_t)(i + ((i >= 8) ? 8 : 0)) * ld];
  return f.v;
}

// ---------- weight transpose/convert: W[K][N] f32 -> Wt[N][K] f16 ----------
__global__ __launch_bounds__(256) void wtrans_kernel(const float* __restrict__ W,
                                                     half_t* __restrict__ Wt,
                                                     int K, int N) {
  int idx = blockIdx.x * 256 + threadIdx.x;
  if (idx >= N * K) return;
  int n = idx / K, k = idx - n * K;
  Wt[idx] = (half_t)W[(size_t)k * N + n];
}

// ---------- LayerNorm over 512 -> f16 (one wave per row) ----------
__global__ __launch_bounds__(256) void ln512_kernel(const float* __restrict__ in, int ldin,
                                                    const float* __restrict__ g,
                                                    const float* __restrict__ bta,
                                                    half_t* __restrict__ out, int rows) {
  int row = blockIdx.x * 8 + (threadIdx.x >> 5);
  if (row >= rows) return;
  int lane = threadIdx.x & 31;
  const float* rp = in + (size_t)row * ldin;
  float x[16];
  float s = 0.0f;
#pragma unroll
  for (int i = 0; i < 16; ++i) { x[i] = rp[lane + 32 * i]; s += x[i]; }
  s = redsum32(s);
  float mean = s * (1.0f / 512.0f);
  float v = 0.0f;
#pragma unroll
  for (int i = 0; i < 16; ++i) { float d = x[i] - mean; v += d * d; }
  v = redsum32(v) * (1.0f / 512.0f);
  float rstd = rsqrtf(v + 1e-5f);
#pragma unroll
  for (int i = 0; i < 16; ++i) {
    int c = lane + 32 * i;
    out[(size_t)row * 512 + c] = (half_t)((x[i] - mean) * rstd * g[c] + bta[c]);
  }
}

// ---------- pipelined 128x256 WMMA GEMM ----------
// C = A[M,K]f16 x Wt[N,K]^T + bias (+res)(relu); wave tile 64x64 (4x4 frags),
// double-buffered LDS, async global->LDS DMA for step i+1 in flight during
// step i math; one barrier per K-step.
template<int HAS_RES, int RELU, int OUT_F>
__global__ __launch_bounds__(256) void gemm_kernel(const half_t* __restrict__ A, int lda,
                                                   const half_t* __restrict__ Wt,
                                                   const float* __restrict__ bias,
                                                   const float* __restrict__ res, int ldr,
                                                   float* __restrict__ outF,
                                                   half_t* __restrict__ outB, int ldo,
                                                   int K) {
  __shared__ __align__(16) half_t As[2][128 * 32];
  __shared__ __align__(16) half_t Bs[2][256 * 32];
  const int tid = threadIdx.x;
  const int lane = tid & 31, wv = tid >> 5;
  const int wm = wv >> 2, wn = wv & 3;    // 2 (M) x 4 (N) waves, 64x64 each
  const int rb = blockIdx.y * 128;
  const int cb = blockIdx.x * 256;

  const int arow = tid >> 1, acol = (tid & 1) * 16;
  const half_t* aptr = A + (size_t)(rb + arow) * lda + acol;
  const half_t* bptr = Wt + (size_t)(cb + tid) * K;   // one N-row per thread

  auto stage = [&](int k0, int buf) {
    uint32_t al = lds_addr(&As[buf][arow * 32 + acol]);
    const half_t* ag = aptr + k0;
    async_b128(al, ag);
    async_b128_off16(al, ag);
    uint32_t bl = lds_addr(&Bs[buf][tid * 32]);
    const half_t* bg = bptr + k0;
    async_b128(bl, bg);
    async_b128_off16(bl, bg);
    async_b128_off32(bl, bg);
    async_b128_off48(bl, bg);
  };

  v8f acc[4][4];
#pragma unroll
  for (int mt = 0; mt < 4; ++mt)
#pragma unroll
    for (int nt = 0; nt < 4; ++nt) acc[mt][nt] = zero8();

  const int steps = K >> 5;
  stage(0, 0);
  wait_async0();
  __syncthreads();
  for (int i = 0; i < steps; ++i) {
    const int buf = i & 1;
    if (i + 1 < steps) stage((i + 1) << 5, buf ^ 1);  // DMA overlaps math below
    v16h af[4], bf[4];
#pragma unroll
    for (int mt = 0; mt < 4; ++mt)
      af[mt] = load_a_frag(&As[buf][(wm * 64 + mt * 16) * 32], 32);
#pragma unroll
    for (int nt = 0; nt < 4; ++nt)
      bf[nt] = load_b_frag_nk(&Bs[buf][(wn * 64 + nt * 16) * 32], 32);
#pragma unroll
    for (int mt = 0; mt < 4; ++mt)
#pragma unroll
      for (int nt = 0; nt < 4; ++nt)
        acc[mt][nt] = __builtin_amdgcn_wmma_f32_16x16x32_f16(
            false, af[mt], false, bf[nt], (short)0, acc[mt][nt], false, false);
    wait_async0();
    __syncthreads();
  }

  const int hi8 = (lane >= 16) ? 8 : 0;
#pragma unroll
  for (int nt = 0; nt < 4; ++nt) {
    int col = cb + wn * 64 + nt * 16 + (lane & 15);
    float bv = bias[col];
#pragma unroll
    for (int mt = 0; mt < 4; ++mt) {
#pragma unroll
      for (int r = 0; r < 8; ++r) {
        int row = rb + wm * 64 + mt * 16 + r + hi8;
        float val = acc[mt][nt][r] + bv;
        if (HAS_RES) val += res[(size_t)row * ldr + col];
        if (RELU)    val = fmaxf(val, 0.0f);
        if (OUT_F)   outF[(size_t)row * ldo + col] = val;
        else         outB[(size_t)row * ldo + col] = (half_t)val;
      }
    }
  }
}

// ---------- dilated attention (flash-style, 512 keys per segment) ----------
__global__ __launch_bounds__(256) void attn_kernel(const half_t* __restrict__ Q,
                                                   const half_t* __restrict__ Kg,
                                                   const half_t* __restrict__ V,
                                                   half_t* __restrict__ O) {
  __shared__ __align__(16) half_t kt[64 * 64];
  __shared__ __align__(16) half_t vt[64 * 64];
  __shared__ __align__(16) half_t pb[8][32 * 64];

  int bid = blockIdx.x;
  int s, rr, off, h0, seg, bat, hg, qh;
  if (bid < 512) { s = 512;  rr = 1; off = 0; h0 = 0;
    qh = bid & 1; hg = (bid >> 1) & 3; seg = (bid >> 3) & 15; bat = bid >> 7;
  } else { int t = bid - 512; s = 1024; rr = 2; off = 1; h0 = 4;
    qh = t & 1; hg = (t >> 1) & 3; seg = (t >> 3) & 7; bat = t >> 6;
  }
  const int head = h0 + hg;
  const int lane = threadIdx.x & 31, wv = threadIdx.x >> 5;
  const int hi = (lane >= 16) ? 1 : 0;
  const size_t bbase = (size_t)bat * SEQ * D;
  const int segtok = seg * s + off;
  const int qrow0 = qh * 256 + wv * 32;
  const float scale = 0.125f;   // 1/sqrt(64)

  // Q fragments: 2 M-tiles x 2 feature K-steps, kept resident
  v16h qf[2][2];
#pragma unroll
  for (int mt = 0; mt < 2; ++mt) {
    int qlocal = qrow0 + mt * 16 + (lane & 15);
    int tok = segtok + rr * qlocal;
    const half_t* rp = Q + bbase + (size_t)tok * D + head * DH + hi * 8;
#pragma unroll
    for (int kf = 0; kf < 2; ++kf) {
      union { v16h v; uint32_t u[8]; } f;
#pragma unroll
      for (int j = 0; j < 8; ++j) {
        int c = kf * 32 + 2 * j + ((2 * j >= 8) ? 8 : 0);
        f.u[j] = *(const uint32_t*)(rp + c);
      }
      qf[mt][kf] = f.v;
    }
  }

  v8f o[2][4];
  float mstat[2][8], lstat[2][8];
#pragma unroll
  for (int mt = 0; mt < 2; ++mt) {
#pragma unroll
    for (int nt = 0; nt < 4; ++nt) o[mt][nt] = zero8();
#pragma unroll
    for (int r = 0; r < 8; ++r) { mstat[mt][r] = -3.0e38f; lstat[mt][r] = 0.0f; }
  }

  for (int kb = 0; kb < 8; ++kb) {
    __syncthreads();
    {  // async-DMA 64 keys of K and V into LDS
      int t = threadIdx.x;
      int kr = t >> 2, dsg = (t & 3) * 16;
      int tok = segtok + rr * (kb * 64 + kr);
      const half_t* ks = Kg + bbase + (size_t)tok * D + head * DH + dsg;
      uint32_t kd = lds_addr(&kt[kr * 64 + dsg]);
      async_b128(kd, ks);
      async_b128_off16(kd, ks);
      const half_t* vs = V + bbase + (size_t)tok * D + head * DH + dsg;
      uint32_t vd = lds_addr(&vt[kr * 64 + dsg]);
      async_b128(vd, vs);
      async_b128_off16(vd, vs);
      wait_async0();
    }
    __syncthreads();

    // scores S = Q K^T
    v8f sc[2][4];
#pragma unroll
    for (int mt = 0; mt < 2; ++mt)
#pragma unroll
      for (int nt = 0; nt < 4; ++nt) sc[mt][nt] = zero8();
#pragma unroll
    for (int kf = 0; kf < 2; ++kf) {
      v16h bf[4];
#pragma unroll
      for (int nt = 0; nt < 4; ++nt)
        bf[nt] = load_b_frag_nk(kt + (nt * 16) * 64 + kf * 32, 64);
#pragma unroll
      for (int mt = 0; mt < 2; ++mt)
#pragma unroll
        for (int nt = 0; nt < 4; ++nt)
          sc[mt][nt] = __builtin_amdgcn_wmma_f32_16x16x32_f16(
              false, qf[mt][kf], false, bf[nt], (short)0, sc[mt][nt], false, false);
    }
#pragma unroll
    for (int mt = 0; mt < 2; ++mt)
#pragma unroll
      for (int nt = 0; nt < 4; ++nt)
#pragma unroll
        for (int e = 0; e < 8; ++e) sc[mt][nt][e] *= scale;

    // online softmax
#pragma unroll
    for (int mt = 0; mt < 2; ++mt) {
#pragma unroll
      for (int r = 0; r < 8; ++r) {
        float rm = -3.0e38f;
#pragma unroll
        for (int nt = 0; nt < 4; ++nt) rm = fmaxf(rm, sc[mt][nt][r]);
        rm = redmax16(rm);
        float mnew = fmaxf(mstat[mt][r], rm);
        float alpha = __expf(mstat[mt][r] - mnew);
        mstat[mt][r] = mnew;
        float rs = 0.0f;
#pragma unroll
        for (int nt = 0; nt < 4; ++nt) {
          float pe = __expf(sc[mt][nt][r] - mnew);
          sc[mt][nt][r] = pe;
          rs += pe;
        }
        rs = redsum16(rs);
        lstat[mt][r] = lstat[mt][r] * alpha + rs;
#pragma unroll
        for (int nt = 0; nt < 4; ++nt) o[mt][nt][r] *= alpha;
      }
    }

    // P -> LDS (wave-private), then O += P x V
#pragma unroll
    for (int mt = 0; mt < 2; ++mt)
#pragma unroll
      for (int nt = 0; nt < 4; ++nt)
#pragma unroll
        for (int r = 0; r < 8; ++r)
          pb[wv][(mt * 16 + r + hi * 8) * 64 + nt * 16 + (lane & 15)] = (half_t)sc[mt][nt][r];

#pragma unroll
    for (int kk = 0; kk < 2; ++kk) {
      v16h vf[4], pf[2];
#pragma unroll
      for (int nt = 0; nt < 4; ++nt)
        vf[nt] = load_b_frag_kn(vt + (size_t)(kk * 32) * 64 + nt * 16, 64);
#pragma unroll
      for (int mt = 0; mt < 2; ++mt)
        pf[mt] = load_a_frag(pb[wv] + mt * 16 * 64 + kk * 32, 64);
#pragma unroll
      for (int mt = 0; mt < 2; ++mt)
#pragma unroll
        for (int nt = 0; nt < 4; ++nt)
          o[mt][nt] = __builtin_amdgcn_wmma_f32_16x16x32_f16(
              false, pf[mt], false, vf[nt], (short)0, o[mt][nt], false, false);
    }
  }

  // finalize and scatter to gathered token positions
#pragma unroll
  for (int mt = 0; mt < 2; ++mt) {
#pragma unroll
    for (int r = 0; r < 8; ++r) {
      int qlocal = qrow0 + mt * 16 + r + hi * 8;
      int tok = segtok + rr * qlocal;
      float inv = 1.0f / lstat[mt][r];
#pragma unroll
      for (int nt = 0; nt < 4; ++nt)
        O[bbase + (size_t)tok * D + head * DH + nt * 16 + (lane & 15)] =
            (half_t)(o[mt][nt][r] * inv);
    }
  }
}

// ---------- host launch ----------
extern "C" void kernel_launch(void* const* d_in, const int* in_sizes, int n_in,
                              void* d_out, int out_size, void* d_ws, size_t ws_size,
                              hipStream_t stream) {
  (void)in_sizes; (void)n_in; (void)out_size; (void)ws_size;
  const float* x    = (const float*)d_in[0];
  const float* ln1g = (const float*)d_in[1];
  const float* ln1b = (const float*)d_in[2];
  const float* Wq = (const float*)d_in[3];  const float* bq = (const float*)d_in[4];
  const float* Wk = (const float*)d_in[5];  const float* bk = (const float*)d_in[6];
  const float* Wv = (const float*)d_in[7];  const float* bv = (const float*)d_in[8];
  const float* Wo = (const float*)d_in[9];  const float* bo = (const float*)d_in[10];
  const float* ln2g = (const float*)d_in[11]; const float* ln2b = (const float*)d_in[12];
  const float* W1 = (const float*)d_in[13]; const float* b1 = (const float*)d_in[14];
  const float* W2 = (const float*)d_in[15]; const float* b2 = (const float*)d_in[16];
  float* out = (float*)d_out;

  char* p = (char*)d_ws;
  auto take = [&](size_t bytes) -> char* {
    char* q = p; p += (bytes + 255) & ~(size_t)255; return q;
  };
  const size_t M = (size_t)MROWS;
  half_t* xln = (half_t*)take(M * 512 * 2);
  half_t* qb  = (half_t*)take(M * 512 * 2);
  half_t* kb  = (half_t*)take(M * 512 * 2);
  half_t* vb  = (half_t*)take(M * 512 * 2);
  half_t* at  = (half_t*)take(M * 512 * 2);
  half_t* h2  = (half_t*)take(M * 512 * 2);
  half_t* a1  = (half_t*)take(M * 1024 * 2);
  half_t* WqT = (half_t*)take(512 * 512 * 2);
  half_t* WkT = (half_t*)take(512 * 512 * 2);
  half_t* WvT = (half_t*)take(512 * 512 * 2);
  half_t* WoT = (half_t*)take(512 * 512 * 2);
  half_t* W1T = (half_t*)take(512 * 1024 * 2);
  half_t* W2T = (half_t*)take(1024 * 512 * 2);

  dim3 blk(256);
  wtrans_kernel<<<(512 * 512 + 255) / 256, blk, 0, stream>>>(Wq, WqT, 512, 512);
  wtrans_kernel<<<(512 * 512 + 255) / 256, blk, 0, stream>>>(Wk, WkT, 512, 512);
  wtrans_kernel<<<(512 * 512 + 255) / 256, blk, 0, stream>>>(Wv, WvT, 512, 512);
  wtrans_kernel<<<(512 * 512 + 255) / 256, blk, 0, stream>>>(Wo, WoT, 512, 512);
  wtrans_kernel<<<(512 * 1024 + 255) / 256, blk, 0, stream>>>(W1, W1T, 512, 1024);
  wtrans_kernel<<<(1024 * 512 + 255) / 256, blk, 0, stream>>>(W2, W2T, 1024, 512);

  // LN1(x2) -> f16
  ln512_kernel<<<MROWS / 8, blk, 0, stream>>>(x + 512, 1024, ln1g, ln1b, xln, MROWS);

  // attention output buffer must be zero (config-1 even positions stay zero)
  hipMemsetAsync(at, 0, M * 512 * 2, stream);

  // QKV projections (f16 out)
  gemm_kernel<0,0,0><<<dim3(2, 256), blk, 0, stream>>>(xln, 512, WqT, bq, nullptr, 0, nullptr, qb, 512, 512);
  gemm_kernel<0,0,0><<<dim3(2, 256), blk, 0, stream>>>(xln, 512, WkT, bk, nullptr, 0, nullptr, kb, 512, 512);
  gemm_kernel<0,0,0><<<dim3(2, 256), blk, 0, stream>>>(xln, 512, WvT, bv, nullptr, 0, nullptr, vb, 512, 512);

  // dilated attention: 512 blocks (cfg0) + 256 blocks (cfg1)
  attn_kernel<<<768, blk, 0, stream>>>(qb, kb, vb, at);

  // y1 = x1 + attn @ Wo + bo  -> out[..., :512] (fp32)
  gemm_kernel<1,0,1><<<dim3(2, 256), blk, 0, stream>>>(at, 512, WoT, bo, x, 1024, out, nullptr, 1024, 512);

  // h = LN2(y1) -> f16
  ln512_kernel<<<MROWS / 8, blk, 0, stream>>>(out, 1024, ln2g, ln2b, h2, MROWS);

  // a1 = relu(h @ W1 + b1) -> f16
  gemm_kernel<0,1,0><<<dim3(4, 256), blk, 0, stream>>>(h2, 512, W1T, b1, nullptr, 0, nullptr, a1, 1024, 512);

  // y2 = x2 + a1 @ W2 + b2 -> out[..., 512:]
  gemm_kernel<1,0,1><<<dim3(2, 256), blk, 0, stream>>>(a1, 1024, W2T, b2, x + 512, 1024, out + 512, nullptr, 1024, 1024);
}
